// FunctionModel_54348516163883
// MI455X (gfx1250) — compile-verified, matrix operand order checked by reading, not run on hardware
//
#include <hip/hip_runtime.h>
#include <hip/hip_bf16.h>

typedef __attribute__((ext_vector_type(16))) _Float16 v16h;
typedef __attribute__((ext_vector_type(8)))  _Float16 v8h;
typedef __attribute__((ext_vector_type(8)))  float    v8f;

#define HSTR   264            // staged row stride in halves: 256 + 8 pad, 528B = 33*16B
#define NFRAG  128            // 16 n-tiles * 8 k-steps
#define WELTS  (NFRAG*32*16)  // 65536 halves = 128KB of f16 weights

// ---------------------------------------------------------------------------
// Pre-pass: WH (256x256 fp32, row-major [out][in]) -> f16 in fragment-linear
// layout matching the CDNA5 16-bit B-matrix (32x16) lane layout:
//   frag f = ntile*8 + kstep ; lane l: column n = 16*ntile + (l&15), g = l>>4
//   halves j=0..7  -> K = 32*kstep + 8*g + j
//   halves j=8..15 -> K = 32*kstep + 16 + 8*g + (j-8)
// Since out[m][n] = sum_k h[m][k] * WH[n][k], B column n == WH row n directly.
// ---------------------------------------------------------------------------
__global__ void prep_wh_kernel(const float* __restrict__ WH, _Float16* __restrict__ whf) {
    int idx = blockIdx.x * blockDim.x + threadIdx.x;   // one thread per (frag,lane)
    if (idx >= NFRAG * 32) return;
    int f = idx >> 5, lane = idx & 31;
    int nt = f >> 3, ks = f & 7;
    int row = nt * 16 + (lane & 15);
    int g   = lane >> 4;
    int kb  = ks * 32 + g * 8;
    _Float16* dst = whf + (size_t)idx * 16;
    #pragma unroll
    for (int j = 0; j < 8; ++j) dst[j]     = (_Float16)WH[row * 256 + kb + j];
    #pragma unroll
    for (int j = 0; j < 8; ++j) dst[8 + j] = (_Float16)WH[row * 256 + kb + 16 + j];
}

// ---------------------------------------------------------------------------
// Async cache->LDS DMA stage of the 128KB f16 weight image (no VGPR data path,
// tracked by ASYNCcnt). Each wave instruction moves 32 lanes x 16B = 512B.
// ---------------------------------------------------------------------------
__device__ __forceinline__ void stage_weights_async(const _Float16* __restrict__ whf,
                                                    _Float16* wbuf, int tid) {
    unsigned lds_base = (unsigned)(size_t)wbuf;   // ISA 10.2: LDS addr = flat addr[31:0]
    #pragma unroll 4
    for (int i = tid; i < (WELTS * 2) / 16; i += 128) {
        unsigned boff = (unsigned)i * 16u;
        asm volatile("global_load_async_to_lds_b128 %0, %1, %2"
                     :: "v"(lds_base + boff), "v"(boff), "s"(whf)
                     : "memory");
    }
    asm volatile("s_wait_asynccnt 0x0" ::: "memory");
}

// ---------------------------------------------------------------------------
// One hidden layer for a 16-row tile: h <- relu(h @ WH^T + bH), h in LDS f16.
// 4 waves; wave w computes n-tiles {w, w+4, w+8, w+12}. A fragments for the
// whole K=256 are register-resident; B fragments stream from LDS weights.
// ---------------------------------------------------------------------------
__device__ __forceinline__ void hidden_layer(_Float16* hbuf, const _Float16* wbuf,
                                             const float* __restrict__ bH,
                                             int wave, int lane) {
    int m = lane & 15, g = lane >> 4;
    // Load all 8 A fragments (16x32 f16 each) for this wave's 16 rows.
    v16h a[8];
    const _Float16* rp = hbuf + m * HSTR + g * 8;
    #pragma unroll
    for (int ks = 0; ks < 8; ++ks) {
        v8h lo = *(const v8h*)(rp + ks * 32);        // K = 32ks + 8g .. +7
        v8h hi = *(const v8h*)(rp + ks * 32 + 16);   // K = 32ks + 16 + 8g .. +7
        a[ks] = __builtin_shufflevector(lo, hi, 0,1,2,3,4,5,6,7,8,9,10,11,12,13,14,15);
    }
    __syncthreads();   // everyone captured h before we overwrite it below

    const v16h* wb = (const v16h*)wbuf;
    #pragma unroll
    for (int i = 0; i < 4; ++i) {
        int nt = wave + i * 4;
        float bv = bH[nt * 16 + m];
        v8f c = {bv, bv, bv, bv, bv, bv, bv, bv};
        #pragma unroll
        for (int ks = 0; ks < 8; ++ks) {
            v16h b = wb[(nt * 8 + ks) * 32 + lane];
            c = __builtin_amdgcn_wmma_f32_16x16x32_f16(
                    false, a[ks], false, b, (short)0, c, false, false);
        }
        // ReLU, downconvert, scatter back: D elem r -> (row = r + 8g, col = 16nt + m)
        #pragma unroll
        for (int r = 0; r < 8; ++r) {
            float v = c[r] > 0.f ? c[r] : 0.f;
            hbuf[(r + 8 * g) * HSTR + nt * 16 + m] = (_Float16)v;
        }
    }
    __syncthreads();
}

// Input layer h = [x,y] @ WI^T + bI for 16 rows; thread: row tid>>3, 32 cols.
__device__ __forceinline__ void input_layer(_Float16* hbuf, float xv, float yv,
                                            const float* __restrict__ WI,
                                            const float* __restrict__ bI, int tid) {
    int m = tid >> 3, c0 = (tid & 7) * 32;
    #pragma unroll 4
    for (int n = c0; n < c0 + 32; ++n) {
        float v = fmaf(xv, WI[2 * n], fmaf(yv, WI[2 * n + 1], bI[n]));
        hbuf[m * HSTR + n] = (_Float16)v;
    }
    __syncthreads();
}

// Output layer: per-row dot(h, WO) via 8 lanes/row + shfl_xor reduce.
__device__ __forceinline__ float output_layer(const _Float16* hbuf,
                                              const float* __restrict__ WO, int tid) {
    int m = tid >> 3, k0 = (tid & 7) * 32;
    float acc = 0.f;
    #pragma unroll 4
    for (int k = k0; k < k0 + 32; ++k)
        acc = fmaf((float)hbuf[m * HSTR + k], WO[k], acc);
    acc += __shfl_xor(acc, 1);
    acc += __shfl_xor(acc, 2);
    acc += __shfl_xor(acc, 4);
    return acc;   // valid in all 8 lanes of the row group
}

// ---------------------------------------------------------------------------
// Interpolation: 32768 independent 16-row tiles through the MLP.
// ---------------------------------------------------------------------------
__global__ void __launch_bounds__(128)
interp_kernel(const float* __restrict__ x, const float* __restrict__ y,
              const _Float16* __restrict__ whf,
              const float* __restrict__ WI, const float* __restrict__ bI,
              const float* __restrict__ bH, const float* __restrict__ WO,
              const float* __restrict__ bO, float* __restrict__ out, int ntiles) {
    __shared__ __attribute__((aligned(16))) _Float16 wbuf[WELTS];      // 128KB
    __shared__ __attribute__((aligned(16))) _Float16 hbuf[16 * HSTR];  // 8.25KB
    int tid = threadIdx.x;
    stage_weights_async(whf, wbuf, tid);
    __syncthreads();

    int wave = tid >> 5, lane = tid & 31;
    float bOv = bO[0];

    for (int tile = blockIdx.x; tile < ntiles; tile += gridDim.x) {
        int rbase = tile * 16;
        int m = tid >> 3;
        // prefetch next tile's activations into cache while this tile computes
        int nxt = tile + gridDim.x;
        if (nxt < ntiles) {
            __builtin_prefetch(&x[nxt * 16 + m], 0, 0);
            __builtin_prefetch(&y[nxt * 16 + m], 0, 0);
        }
        input_layer(hbuf, x[rbase + m], y[rbase + m], WI, bI, tid);
        hidden_layer(hbuf, wbuf, bH, wave, lane);
        hidden_layer(hbuf, wbuf, bH, wave, lane);
        float acc = output_layer(hbuf, WO, tid);
        if ((tid & 7) == 0) {
            int row = rbase + (tid >> 3);
            int b = row >> 11, s = row & 2047;     // row = b*2048 + s
            out[b * 2560 + s] = acc + bOv;
        }
        __syncthreads();   // outputs read hbuf; next tile overwrites it
    }
}

// ---------------------------------------------------------------------------
// Extrapolation: 256 independent per-batch recurrences -> 16 blocks of
// 16 batches each; each block runs its own 512-step chain (one WMMA M-tile).
// ---------------------------------------------------------------------------
__global__ void __launch_bounds__(128)
extrap_kernel(const float* __restrict__ xe, const _Float16* __restrict__ whf,
              const float* __restrict__ WI, const float* __restrict__ bI,
              const float* __restrict__ bH, const float* __restrict__ WO,
              const float* __restrict__ bO, float* __restrict__ out) {
    __shared__ __attribute__((aligned(16))) _Float16 wbuf[WELTS];
    __shared__ __attribute__((aligned(16))) _Float16 hbuf[16 * HSTR];
    __shared__ float ybuf[2][16];
    int tid = threadIdx.x;
    int bb = blockIdx.x * 16;          // batch base for this block
    stage_weights_async(whf, wbuf, tid);
    if (tid < 16) ybuf[0][tid] = out[(bb + tid) * 2560 + 2047];  // y0 = interp[:, -1]
    __syncthreads();

    int wave = tid >> 5, lane = tid & 31;
    float bOv = bO[0];

    for (int t = 0; t < 512; ++t) {
        int m = tid >> 3;
        float xv = xe[(bb + m) * 512 + t];
        float yv = ybuf[t & 1][m];
        input_layer(hbuf, xv, yv, WI, bI, tid);
        hidden_layer(hbuf, wbuf, bH, wave, lane);
        hidden_layer(hbuf, wbuf, bH, wave, lane);
        float acc = output_layer(hbuf, WO, tid);
        if ((tid & 7) == 0) {
            float ynew = acc + bOv;
            ybuf[(t + 1) & 1][tid >> 3] = ynew;
            out[(bb + (tid >> 3)) * 2560 + 2048 + t] = ynew;
        }
        __syncthreads();   // publish ynew, release hbuf for next step
    }
}

// ---------------------------------------------------------------------------
extern "C" void kernel_launch(void* const* d_in, const int* in_sizes, int n_in,
                              void* d_out, int out_size, void* d_ws, size_t ws_size,
                              hipStream_t stream) {
    (void)in_sizes; (void)n_in; (void)out_size; (void)ws_size;
    const float* x_interp = (const float*)d_in[0];   // (256, 2048)
    const float* y_interp = (const float*)d_in[1];   // (256, 2048)
    const float* x_extrap = (const float*)d_in[2];   // (256, 512)
    const float* WI = (const float*)d_in[3];         // (256, 2)
    const float* bI = (const float*)d_in[4];         // (256,)
    const float* WH = (const float*)d_in[5];         // (256, 256)
    const float* bH = (const float*)d_in[6];         // (256,)
    const float* WO = (const float*)d_in[7];         // (1, 256)
    const float* bO = (const float*)d_in[8];         // (1,)
    float* out = (float*)d_out;                      // (256, 2560)
    _Float16* whf = (_Float16*)d_ws;                 // 128KB f16 swizzled WH

    prep_wh_kernel<<<16, 256, 0, stream>>>(WH, whf);
    interp_kernel<<<2048, 128, 0, stream>>>(x_interp, y_interp, whf,
                                            WI, bI, bH, WO, bO, out, 32768);
    extrap_kernel<<<16, 128, 0, stream>>>(x_extrap, whf, WI, bI, bH, WO, bO, out);
}
